// TriangleMultiplicationOutgoing_7215545057295
// MI455X (gfx1250) — matrix operand order repaired and checked by reading, not compiled, not run on hardware
//
#include <hip/hip_runtime.h>
#include <hip/hip_bf16.h>

// ---------------------------------------------------------------------------
// Triangle multiplication (outgoing), N=384, C=M=128, fp32 in/out.
// All GEMMs via v_wmma_f32_16x16x32_f16 (f16 inputs, f32 accumulate).
// Triangle einsum uses async global->LDS staging (ASYNCcnt) + LDS double buffer.
// ---------------------------------------------------------------------------

typedef __attribute__((ext_vector_type(16))) _Float16 v16h;
typedef __attribute__((ext_vector_type(8)))  _Float16 v8h;
typedef __attribute__((ext_vector_type(4)))  _Float16 v4h;
typedef __attribute__((ext_vector_type(8)))  float    v8f;
typedef __attribute__((ext_vector_type(4)))  int      v4i_;

#define NDIM 384
#define NN   (384 * 384)     // 147456
#define CH   128

__device__ __forceinline__ float sigmoidf_(float x) {
    return 1.0f / (1.0f + __expf(-x));
}

// A-matrix fragment, 16x32 f16, source row-major [row][k], row stride ld.
__device__ __forceinline__ v16h load_frag_a(const _Float16* base, int ld, int lane) {
    int row = lane & 15;
    int kb  = (lane >> 4) << 3;
    const _Float16* p = base + (size_t)row * ld + kb;
    v8h lo = *(const v8h*)(p);
    v8h hi = *(const v8h*)(p + 16);
    v16h f;
#pragma unroll
    for (int t = 0; t < 8; ++t) { f[t] = lo[t]; f[t + 8] = hi[t]; }
    return f;
}

// B-matrix fragment, 32x16 f16, B = S^T where S is row-major [col][k], stride ld.
__device__ __forceinline__ v16h load_frag_b(const _Float16* base, int ld, int lane) {
    int col = lane & 15;
    int kb  = (lane >> 4) << 3;
    const _Float16* p = base + (size_t)col * ld + kb;
    v8h lo = *(const v8h*)(p);
    v8h hi = *(const v8h*)(p + 16);
    v16h f;
#pragma unroll
    for (int t = 0; t < 8; ++t) { f[t] = lo[t]; f[t + 8] = hi[t]; }
    return f;
}

__device__ __forceinline__ v8f wmma16(v16h a, v16h b, v8f c) {
    return __builtin_amdgcn_wmma_f32_16x16x32_f16(
        false, a, false, b, (short)0, c, false, false);
}

// ---- CDNA5 async global->LDS copy (16B per lane), ASYNCcnt-tracked ---------
__device__ __forceinline__ void async_copy16(const _Float16* g, _Float16* l) {
#if __has_builtin(__builtin_amdgcn_global_load_async_to_lds_b128)
    __builtin_amdgcn_global_load_async_to_lds_b128(
        (__attribute__((address_space(1))) v4i_*)(size_t)g,
        (__attribute__((address_space(3))) v4i_*)(unsigned)(size_t)l,
        0, 0);
#else
    unsigned loff = (unsigned)(size_t)l;   // low 32 bits of generic LDS addr = LDS offset
    asm volatile("global_load_async_to_lds_b128 %0, %1, off"
                 :: "v"(loff), "v"(g) : "memory");
#endif
}

__device__ __forceinline__ void wait_async_0() {
#if __has_builtin(__builtin_amdgcn_s_wait_asynccnt)
    __builtin_amdgcn_s_wait_asynccnt(0);
#else
    asm volatile("s_wait_asynccnt 0x0" ::: "memory");
#endif
}
__device__ __forceinline__ void wait_async_3() {
#if __has_builtin(__builtin_amdgcn_s_wait_asynccnt)
    __builtin_amdgcn_s_wait_asynccnt(3);
#else
    asm volatile("s_wait_asynccnt 0x3" ::: "memory");
#endif
}

// ---------------------------------------------------------------------------
// Kernel 0: transpose + f32->f16 convert 6 weight matrices to [out][in]
// order: 0=lp 1=lg 2=rp 3=rg 4=g 5=out
// ---------------------------------------------------------------------------
__global__ void k_prep_weights(const float* __restrict__ W0, const float* __restrict__ W1,
                               const float* __restrict__ W2, const float* __restrict__ W3,
                               const float* __restrict__ W4, const float* __restrict__ W5,
                               _Float16* __restrict__ Wt) {
    int idx = blockIdx.x * blockDim.x + threadIdx.x;
    if (idx >= 6 * CH * CH) return;
    int w = idx >> 14;
    int r = (idx >> 7) & 127;      // cout
    int c = idx & 127;             // cin
    const float* Ws[6] = {W0, W1, W2, W3, W4, W5};
    Wt[idx] = (_Float16)Ws[w][c * CH + r];   // W stored [fan_in][fan_out]
}

// ---------------------------------------------------------------------------
// LayerNorm over C=128, one wave32 per row, f32 in -> f16 out
// ---------------------------------------------------------------------------
__global__ void k_layernorm_f32_f16(const float* __restrict__ x,
                                    const float* __restrict__ w,
                                    const float* __restrict__ b,
                                    _Float16* __restrict__ y, int rows) {
    int wid  = (blockIdx.x * blockDim.x + threadIdx.x) >> 5;
    int lane = threadIdx.x & 31;
    if (wid >= rows) return;
    const float* xr = x + (size_t)wid * CH;
    float4 v = *(const float4*)(xr + lane * 4);
    float s  = v.x + v.y + v.z + v.w;
    float s2 = v.x * v.x + v.y * v.y + v.z * v.z + v.w * v.w;
#pragma unroll
    for (int m = 16; m >= 1; m >>= 1) {
        s  += __shfl_xor(s, m, 32);
        s2 += __shfl_xor(s2, m, 32);
    }
    float mu  = s * (1.0f / CH);
    float var = s2 * (1.0f / CH) - mu * mu;
    float rs  = rsqrtf(var + 1e-5f);
    float4 wv = *(const float4*)(w + lane * 4);
    float4 bv = *(const float4*)(b + lane * 4);
    v4h o;
    o[0] = (_Float16)((v.x - mu) * rs * wv.x + bv.x);
    o[1] = (_Float16)((v.y - mu) * rs * wv.y + bv.y);
    o[2] = (_Float16)((v.z - mu) * rs * wv.z + bv.z);
    o[3] = (_Float16)((v.w - mu) * rs * wv.w + bv.w);
    *(v4h*)(y + (size_t)wid * CH + lane * 4) = o;
}

// ---------------------------------------------------------------------------
// Kernel 2: fused 5-projection GEMM + gating (16 pos x 16 ch per wave, K=128).
// left/right stored CHANNEL-MAJOR f16 [c][pos]; gate stored [pos][c] f16.
// ---------------------------------------------------------------------------
__global__ void k_proj(const _Float16* __restrict__ actln,   // [NN][128]
                       const _Float16* __restrict__ Wt,      // [6][128][128]
                       const float* __restrict__ mask,       // [NN]
                       const float* __restrict__ b_lp, const float* __restrict__ b_lg,
                       const float* __restrict__ b_rp, const float* __restrict__ b_rg,
                       const float* __restrict__ b_g,
                       _Float16* __restrict__ left_t,        // [128][NN]
                       _Float16* __restrict__ right_t,       // [128][NN]
                       _Float16* __restrict__ gate) {        // [NN][128]
    int wid  = (blockIdx.x * blockDim.x + threadIdx.x) >> 5;
    int lane = threadIdx.x & 31;
    int rowtile = wid >> 3;
    int coltile = wid & 7;
    if (rowtile >= NN / 16) return;
    int p0 = rowtile * 16, c0 = coltile * 16;

    v8f a_lp = {}, a_lg = {}, a_rp = {}, a_rg = {}, a_g = {};
    const _Float16* Abase = actln + (size_t)p0 * CH;
#pragma unroll
    for (int kc = 0; kc < 4; ++kc) {
        v16h a = load_frag_a(Abase + kc * 32, CH, lane);
        v16h b0 = load_frag_b(Wt + 0 * 16384 + (size_t)c0 * CH + kc * 32, CH, lane);
        a_lp = wmma16(a, b0, a_lp);
        v16h b1 = load_frag_b(Wt + 1 * 16384 + (size_t)c0 * CH + kc * 32, CH, lane);
        a_lg = wmma16(a, b1, a_lg);
        v16h b2 = load_frag_b(Wt + 2 * 16384 + (size_t)c0 * CH + kc * 32, CH, lane);
        a_rp = wmma16(a, b2, a_rp);
        v16h b3 = load_frag_b(Wt + 3 * 16384 + (size_t)c0 * CH + kc * 32, CH, lane);
        a_rg = wmma16(a, b3, a_rg);
        v16h b4 = load_frag_b(Wt + 4 * 16384 + (size_t)c0 * CH + kc * 32, CH, lane);
        a_g  = wmma16(a, b4, a_g);
    }
    int n = lane & 15, mb = (lane >> 4) << 3;
    int c = c0 + n;
    float blp = b_lp[c], blg = b_lg[c], brp = b_rp[c], brg = b_rg[c], bg = b_g[c];
    v8h lpack, rpack;
#pragma unroll
    for (int v = 0; v < 8; ++v) {
        int p = p0 + mb + v;
        float mval = mask[p];
        float l = mval * (a_lp[v] + blp) * sigmoidf_(a_lg[v] + blg);
        float r = mval * (a_rp[v] + brp) * sigmoidf_(a_rg[v] + brg);
        lpack[v] = (_Float16)l;
        rpack[v] = (_Float16)r;
        gate[(size_t)p * CH + c] = (_Float16)sigmoidf_(a_g[v] + bg);
    }
    *(v8h*)(left_t  + (size_t)c * NN + p0 + mb) = lpack;
    *(v8h*)(right_t + (size_t)c * NN + p0 + mb) = rpack;
}

// ---------------------------------------------------------------------------
// Kernel 3: triangle einsum. Per channel c: Out_c = L_c @ R_c^T (384x384x384).
// Workgroup (256 thr, 8 waves) does a 128x64 tile of one channel.
// Async global->LDS double-buffered staging; each wave: 32x32 via 4 WMMAs/chunk.
// ---------------------------------------------------------------------------
__device__ __forceinline__ void fill_tiles(const _Float16* L, const _Float16* R,
                                           int i0, int j0, int k0,
                                           _Float16* Lb, _Float16* Rb, int tid) {
    // L tile: 128 rows x 32 K = 512 x 16B segments  (2 per thread)
    int r = tid >> 2, seg = (tid & 3) * 8;
    async_copy16(L + (size_t)(i0 + r) * NDIM + k0 + seg, Lb + r * 32 + seg);
    int idx = tid + 256;
    r = idx >> 2; seg = (idx & 3) * 8;
    async_copy16(L + (size_t)(i0 + r) * NDIM + k0 + seg, Lb + r * 32 + seg);
    // R tile: 64 rows x 32 K = 256 x 16B segments   (1 per thread)
    r = tid >> 2; seg = (tid & 3) * 8;
    async_copy16(R + (size_t)(j0 + r) * NDIM + k0 + seg, Rb + r * 32 + seg);
}

__global__ void __launch_bounds__(256)
k_triangle(const _Float16* __restrict__ left_t,   // [128][384][384]
           const _Float16* __restrict__ right_t,  // [128][384][384]
           float* __restrict__ tri) {             // [384][384][128]
    __shared__ __align__(16) _Float16 Lb[2][128 * 32];
    __shared__ __align__(16) _Float16 Rb[2][64 * 32];
    int tid  = threadIdx.x;
    int lane = tid & 31;
    int wave = tid >> 5;               // 0..7
    int wy = wave >> 1, wx = wave & 1; // 4 x 2 wave grid
    int bid = blockIdx.x;              // 128 channels * 3 * 6 tiles
    int c   = bid / 18;
    int t   = bid % 18;
    int i0  = (t / 6) * 128;
    int j0  = (t % 6) * 64;
    const _Float16* L = left_t  + (size_t)c * NN;
    const _Float16* R = right_t + (size_t)c * NN;

    fill_tiles(L, R, i0, j0, 0, Lb[0], Rb[0], tid);

    v8f a00 = {}, a01 = {}, a10 = {}, a11 = {};
    int buf = 0;
    for (int kc = 0; kc < 12; ++kc) {
        if (kc + 1 < 12) {
            fill_tiles(L, R, i0, j0, (kc + 1) * 32, Lb[buf ^ 1], Rb[buf ^ 1], tid);
            wait_async_3();            // previous fill (3 loads) has landed
        } else {
            wait_async_0();
        }
        __syncthreads();               // whole tile visible to all 8 waves

        const _Float16* lb = Lb[buf];
        const _Float16* rb = Rb[buf];
        v16h fa0 = load_frag_a(lb + (wy * 32) * 32,      32, lane);
        v16h fa1 = load_frag_a(lb + (wy * 32 + 16) * 32, 32, lane);
        v16h fb0 = load_frag_b(rb + (wx * 32) * 32,      32, lane);
        v16h fb1 = load_frag_b(rb + (wx * 32 + 16) * 32, 32, lane);
        a00 = wmma16(fa0, fb0, a00);
        a01 = wmma16(fa0, fb1, a01);
        a10 = wmma16(fa1, fb0, a10);
        a11 = wmma16(fa1, fb1, a11);

        __syncthreads();               // done reading buf before it is refilled
        buf ^= 1;
    }

    int n = lane & 15, mb = (lane >> 4) << 3;
    int ibase = i0 + wy * 32 + mb;
    int jbase = j0 + wx * 32 + n;
#pragma unroll
    for (int v = 0; v < 8; ++v) {
        int i = ibase + v;
        tri[((size_t)i * NDIM + jbase) * CH + c]              = a00[v];
        tri[((size_t)i * NDIM + jbase + 16) * CH + c]         = a01[v];
        tri[((size_t)(i + 16) * NDIM + jbase) * CH + c]       = a10[v];
        tri[((size_t)(i + 16) * NDIM + jbase + 16) * CH + c]  = a11[v];
    }
}

// ---------------------------------------------------------------------------
// Kernel 4: output projection: (LN(tri) @ W_out + b_out) * gate -> f32 out
// ---------------------------------------------------------------------------
__global__ void k_outproj(const _Float16* __restrict__ triln,   // [NN][128]
                          const _Float16* __restrict__ Wt_out,  // [128][128] [out][in]
                          const float* __restrict__ b_out,
                          const _Float16* __restrict__ gate,    // [NN][128]
                          float* __restrict__ out) {            // [NN][128]
    int wid  = (blockIdx.x * blockDim.x + threadIdx.x) >> 5;
    int lane = threadIdx.x & 31;
    int rowtile = wid >> 3;
    int coltile = wid & 7;
    if (rowtile >= NN / 16) return;
    int p0 = rowtile * 16, c0 = coltile * 16;
    v8f acc = {};
#pragma unroll
    for (int kc = 0; kc < 4; ++kc) {
        v16h a = load_frag_a(triln + (size_t)p0 * CH + kc * 32, CH, lane);
        v16h b = load_frag_b(Wt_out + (size_t)c0 * CH + kc * 32, CH, lane);
        acc = wmma16(a, b, acc);
    }
    int n = lane & 15, mb = (lane >> 4) << 3;
    int c = c0 + n;
    float bo = b_out[c];
#pragma unroll
    for (int v = 0; v < 8; ++v) {
        int p = p0 + mb + v;
        float g = (float)gate[(size_t)p * CH + c];
        out[(size_t)p * CH + c] = (acc[v] + bo) * g;
    }
}

// ---------------------------------------------------------------------------
extern "C" void kernel_launch(void* const* d_in, const int* in_sizes, int n_in,
                              void* d_out, int out_size, void* d_ws, size_t ws_size,
                              hipStream_t stream) {
    (void)in_sizes; (void)n_in; (void)out_size; (void)ws_size;
    const float* act     = (const float*)d_in[0];
    const float* mask    = (const float*)d_in[1];
    const float* ln_in_w = (const float*)d_in[2];
    const float* ln_in_b = (const float*)d_in[3];
    const float* ln_c_w  = (const float*)d_in[4];
    const float* ln_c_b  = (const float*)d_in[5];
    const float* W_lp = (const float*)d_in[6];  const float* b_lp = (const float*)d_in[7];
    const float* W_rp = (const float*)d_in[8];  const float* b_rp = (const float*)d_in[9];
    const float* W_lg = (const float*)d_in[10]; const float* b_lg = (const float*)d_in[11];
    const float* W_rg = (const float*)d_in[12]; const float* b_rg = (const float*)d_in[13];
    const float* W_out = (const float*)d_in[14]; const float* b_out = (const float*)d_in[15];
    const float* W_g  = (const float*)d_in[16]; const float* b_g  = (const float*)d_in[17];
    float* out = (float*)d_out;

    char* ws = (char*)d_ws;
    size_t off = 0;
    auto alloc = [&](size_t bytes) {
        void* p = ws + off;
        off = (off + bytes + 255) & ~(size_t)255;
        return p;
    };
    _Float16* Wt      = (_Float16*)alloc(6 * CH * CH * sizeof(_Float16));
    _Float16* actln   = (_Float16*)alloc((size_t)NN * CH * sizeof(_Float16)); // reused as triln
    _Float16* left_t  = (_Float16*)alloc((size_t)NN * CH * sizeof(_Float16));
    _Float16* right_t = (_Float16*)alloc((size_t)NN * CH * sizeof(_Float16));
    _Float16* gate    = (_Float16*)alloc((size_t)NN * CH * sizeof(_Float16));
    float*    tri     = (float*)   alloc((size_t)NN * CH * sizeof(float));

    k_prep_weights<<<(6 * CH * CH + 255) / 256, 256, 0, stream>>>(
        W_lp, W_lg, W_rp, W_rg, W_g, W_out, Wt);

    k_layernorm_f32_f16<<<NN / 8, 256, 0, stream>>>(act, ln_in_w, ln_in_b, actln, NN);

    k_proj<<<(NN / 16) * 8 / 4, 128, 0, stream>>>(
        actln, Wt, mask, b_lp, b_lg, b_rp, b_rg, b_g, left_t, right_t, gate);

    // 128 channels * (3 i-tiles * 6 j-tiles) workgroups of 256 threads
    k_triangle<<<CH * 18, 256, 0, stream>>>(left_t, right_t, tri);

    k_layernorm_f32_f16<<<NN / 8, 256, 0, stream>>>(tri, ln_c_w, ln_c_b, actln, NN);

    k_outproj<<<(NN / 16) * 8 / 4, 128, 0, stream>>>(actln, Wt + 5 * 16384, b_out, gate, out);
}
